// BeamSplitterAttention_23545010717465
// MI455X (gfx1250) — matrix operand order repaired and checked by reading, not compiled, not run on hardware
//
#include <hip/hip_runtime.h>
#include <cstdint>

// ---------------------------------------------------------------------------
// BeamSplitterAttention for MI455X (gfx1250, wave32, WMMA).
// Compute-bound (~103 GFLOP vs ~100MB traffic) -> all large matmuls via
// v_wmma_f32_16x16x32_bf16; tile staging via async global->LDS (ASYNCcnt)
// with double buffering. Local attention (seq len 4) is VALU.
// ---------------------------------------------------------------------------

typedef __attribute__((ext_vector_type(16))) __bf16 v16bf;
typedef __attribute__((ext_vector_type(8)))  __bf16 v8bf;
typedef __attribute__((ext_vector_type(8)))  float  v8f;
typedef __attribute__((ext_vector_type(4)))  int    v4i;

#define D_MODEL 1024
#define NHEADS  16
#define HD      64
#define BATCH   2
#define SEQ     2048
#define CS      512
#define NC      4
#define NTOK    (BATCH * SEQ)   // 4096 tokens

#if defined(__HIP_DEVICE_COMPILE__) && __has_builtin(__builtin_amdgcn_global_load_async_to_lds_b128)
#define HAVE_ASYNC_LDS 1
#else
#define HAVE_ASYNC_LDS 0
#endif

#if HAVE_ASYNC_LDS
typedef __attribute__((address_space(1))) v4i as1_v4i;  // global int4
typedef __attribute__((address_space(3))) v4i as3_v4i;  // LDS int4
#endif

__device__ __forceinline__ float bf2f(__bf16 x) {
  unsigned short s = __builtin_bit_cast(unsigned short, x);
  unsigned u = ((unsigned)s) << 16;
  return __builtin_bit_cast(float, u);
}
__device__ __forceinline__ __bf16 f2bf(float f) {
  unsigned u = __builtin_bit_cast(unsigned, f);
  u += 0x7fffu + ((u >> 16) & 1u);      // round-to-nearest-even
  unsigned short s = (unsigned short)(u >> 16);
  return __builtin_bit_cast(__bf16, s);
}

union U16 { v16bf v; v8bf h[2]; };

// 16-byte global -> LDS transfer via the gfx1250 async-DMA path (ASYNCcnt).
// AS1 bits == generic bits for global pointers; AS3 offset == low 32 bits of
// a generic LDS pointer.
__device__ __forceinline__ void cp16_to_lds(const void* g, void* l) {
#if HAVE_ASYNC_LDS
  __builtin_amdgcn_global_load_async_to_lds_b128(
      (as1_v4i*)(uintptr_t)g,
      (as3_v4i*)(unsigned)(uintptr_t)l,
      0, 0);
#else
  *(v8bf*)l = *(const v8bf*)g;
#endif
}
__device__ __forceinline__ void async_wait_all() {
#if HAVE_ASYNC_LDS
  asm volatile("s_wait_asynccnt 0" ::: "memory");
#endif
}
// Wait until at most 5 async transfers outstanding (one tile's worth): with
// in-order async-load completion this guarantees the previous tile landed.
__device__ __forceinline__ void async_wait_prev_tile() {
#if HAVE_ASYNC_LDS
  asm volatile("s_wait_asynccnt 5" ::: "memory");
#endif
}

// ---------------------------------------------------------------------------
// Elementwise f32 -> bf16 cast (weights).
// ---------------------------------------------------------------------------
__global__ void cast_f32_bf16(const float* __restrict__ in,
                              __bf16* __restrict__ out, int n) {
  int i = blockIdx.x * blockDim.x + threadIdx.x;
  if (i < n) out[i] = f2bf(in[i]);
}

// ---------------------------------------------------------------------------
// x (B, nc, cs, d) -> x_perm (B, cs, nc, d) flattened, cast to bf16.
// Dest token m = (b*CS + p)*NC + n ; src token = (b*NC + n)*CS + p.
// ---------------------------------------------------------------------------
__global__ void permute_cast_x(const float* __restrict__ x,
                               __bf16* __restrict__ xp) {
  int i = blockIdx.x * blockDim.x + threadIdx.x;   // over NTOK*D_MODEL
  int c  = i & (D_MODEL - 1);
  int m  = i >> 10;
  int n  = m & (NC - 1);
  int bp = m >> 2;
  int p  = bp & (CS - 1);
  int b  = bp >> 9;
  int src = (b * NC + n) * CS + p;
  xp[(size_t)m * D_MODEL + c] = f2bf(x[(size_t)src * D_MODEL + c]);
}

// ---------------------------------------------------------------------------
// WMMA GEMM: C[M,N] = A[M,K] (bf16, row-major) * W[N,K]^T (bf16) + bias.
// Block tile 64x256x32, 256 threads = 8 waves in a 2(m) x 4(n) grid,
// each wave owns 32x64 (2x4 WMMA accumulators, 8 WMMAs per k-step, A frags
// reused across 4 B tiles -> 0.75 LDS fragment loads per WMMA).
// Double-buffered LDS tiles staged with async global->LDS transfers.
// PERM==1 applies the torch "stack/transpose/reshape" row permutation:
//   src row m=(b*cs+p)*nc+n  ->  dst row t=(n*B+b)*cs+p.
// ---------------------------------------------------------------------------
template<int OUT_BF16, int PERM>
__global__ void gemm_bf16(const __bf16* __restrict__ A,
                          const __bf16* __restrict__ W,
                          const float*  __restrict__ bias,
                          void* __restrict__ C,
                          int M, int N, int K) {
  __shared__ __attribute__((aligned(64))) __bf16 As[2][64][32];
  __shared__ __attribute__((aligned(64))) __bf16 Ws[2][256][32];

  const int tid  = threadIdx.x;
  const int lane = tid & 31, wid = tid >> 5;
  const int hf   = lane >> 4, ln = lane & 15;
  const int wm   = wid & 1,   wn = wid >> 1;
  const int bm   = blockIdx.y * 64, bn = blockIdx.x * 256;

  const v8f vz = {0.f,0.f,0.f,0.f,0.f,0.f,0.f,0.f};
  v8f acc[2][4];
  for (int i = 0; i < 2; ++i) for (int j = 0; j < 4; ++j) acc[i][j] = vz;

  const int ar = tid >> 2;            // 0..63
  const int ac = (tid & 3) * 8;       // 0,8,16,24

  // 5 x 16B transfers per thread per k-tile
  auto issue_tile = [&](int k0, int buf) {
    cp16_to_lds(&A[(size_t)(bm + ar) * K + k0 + ac], &As[buf][ar][ac]);
#pragma unroll
    for (int i = 0; i < 4; ++i)
      cp16_to_lds(&W[(size_t)(bn + ar + 64 * i) * K + k0 + ac],
                  &Ws[buf][ar + 64 * i][ac]);
  };

  const int nk = K / 32;
  issue_tile(0, 0);
  for (int kt = 0; kt < nk; ++kt) {
    const int buf = kt & 1;
    if (kt + 1 < nk) {
      issue_tile((kt + 1) * 32, buf ^ 1);   // prefetch next tile (other buffer)
      async_wait_prev_tile();               // current tile resident
    } else {
      async_wait_all();
    }
    __syncthreads();

    // A fragments (16x32 bf16, lanes0-15 K=0..15,
    // lanes16-31 K=16..31)
    v16bf af[2];
#pragma unroll
    for (int mt = 0; mt < 2; ++mt) {
      int m = wm * 32 + mt * 16 + ln;
      U16 u;
      u.h[0] = *(const v8bf*)&As[buf][m][16 * hf];
      u.h[1] = *(const v8bf*)&As[buf][m][16 * hf + 8];
      af[mt] = u.v;
    }
    // B fragments (32x16 bf16: lane = column n, halves split K 0..15/16..31)
#pragma unroll
    for (int nt = 0; nt < 4; ++nt) {
      int n = wn * 64 + nt * 16 + ln;
      v16bf bfr = *(const v16bf*)&Ws[buf][n][16 * hf];
#pragma unroll
      for (int mt = 0; mt < 2; ++mt)
        acc[mt][nt] = __builtin_amdgcn_wmma_f32_16x16x32_bf16(
            false, af[mt], false, bfr, (short)0, acc[mt][nt], false, false);
    }
    __syncthreads();   // protect buffer reuse
  }

  // epilogue: C layout row M = r + 8*hf, col N = ln within 16-tile
  for (int mt = 0; mt < 2; ++mt)
    for (int nt = 0; nt < 4; ++nt) {
      int col = bn + wn * 64 + nt * 16 + ln;
      float bi = bias[col];
      for (int r = 0; r < 8; ++r) {
        int row = bm + wm * 32 + mt * 16 + 8 * hf + r;
        float val = acc[mt][nt][r] + bi;
        int dr = row;
        if (PERM) {
          int n = row & 3, bp = row >> 2, b = bp >> 9, p = bp & 511;
          dr = ((n * BATCH + b) << 9) | p;
        }
        if (OUT_BF16) ((__bf16*)C)[(size_t)dr * N + col] = f2bf(val);
        else          ((float*)C)[(size_t)dr * N + col]  = val;
      }
    }
}

// ---------------------------------------------------------------------------
// Local attention: seq length nc=4 per (b,p,h) -> plain VALU, one thread each.
// qkv rows in (B, cs, nc) order, 3072 cols (q|k|v).
// ---------------------------------------------------------------------------
__global__ void local_attn(const __bf16* __restrict__ qkv,
                           __bf16* __restrict__ out) {
  int t  = blockIdx.x * blockDim.x + threadIdx.x;  // BATCH*CS*NHEADS = 16384
  int h  = t & 15;
  int bp = t >> 4;
  const size_t rb = (size_t)bp * NC;

  float sc[NC][NC];
  for (int qi = 0; qi < NC; ++qi) {
    const __bf16* q = qkv + (rb + qi) * 3072 + h * HD;
    for (int ki = 0; ki < NC; ++ki) {
      const __bf16* k = qkv + (rb + ki) * 3072 + 1024 + h * HD;
      float s = 0.f;
      for (int c = 0; c < HD; ++c) s += bf2f(q[c]) * bf2f(k[c]);
      sc[qi][ki] = s * 0.125f;   // 1/sqrt(64)
    }
  }
  for (int qi = 0; qi < NC; ++qi) {
    float mx = sc[qi][0];
    for (int ki = 1; ki < NC; ++ki) mx = fmaxf(mx, sc[qi][ki]);
    float sum = 0.f;
    for (int ki = 0; ki < NC; ++ki) { sc[qi][ki] = __expf(sc[qi][ki] - mx); sum += sc[qi][ki]; }
    float inv = 1.f / sum;
    for (int ki = 0; ki < NC; ++ki) sc[qi][ki] *= inv;
  }
  for (int qi = 0; qi < NC; ++qi) {
    __bf16* o = out + (rb + qi) * D_MODEL + h * HD;
    for (int c = 0; c < HD; ++c) {
      float v = 0.f;
      for (int ki = 0; ki < NC; ++ki)
        v += sc[qi][ki] * bf2f(qkv[(rb + ki) * 3072 + 2048 + h * HD + c]);
      o[c] = f2bf(v);
    }
  }
}

// ---------------------------------------------------------------------------
// Global flash attention with WMMA. grid = (S/64, B*H), block = 128 (4 waves).
// Wave w handles 16 q-rows; KV streamed in 64-key LDS tiles (K tile async).
// ---------------------------------------------------------------------------
__global__ void global_attn(const __bf16* __restrict__ qkv,
                            __bf16* __restrict__ out) {
  __shared__ __attribute__((aligned(64))) __bf16 Kt[64][64];     // [key][dim]
  __shared__ __attribute__((aligned(64))) __bf16 Vt[64][64];     // [dim][key] (transposed)
  __shared__ __attribute__((aligned(64))) __bf16 Pt[4][16][64];  // per-wave P tile

  const int tid  = threadIdx.x;
  const int lane = tid & 31, w = tid >> 5;
  const int hf   = lane >> 4, ln = lane & 15;
  const int bh = blockIdx.y, b = bh >> 4, h = bh & 15;
  const __bf16* base = qkv + (size_t)b * SEQ * 3072;
  const int q0 = blockIdx.x * 64 + w * 16;

  // Q fragments (held in registers for the whole pass)
  v16bf qf[2];
  {
    const __bf16* qp = base + (size_t)(q0 + ln) * 3072 + h * HD;
    for (int kc = 0; kc < 2; ++kc) {
      U16 u;
      u.h[0] = *(const v8bf*)&qp[kc * 32 + 16 * hf];
      u.h[1] = *(const v8bf*)&qp[kc * 32 + 16 * hf + 8];
      qf[kc] = u.v;
    }
  }

  const v8f vz = {0.f,0.f,0.f,0.f,0.f,0.f,0.f,0.f};
  v8f o[4];
  for (int j = 0; j < 4; ++j) o[j] = vz;
  float mrow[8], lrow[8];
  for (int r = 0; r < 8; ++r) { mrow[r] = -3.0e38f; lrow[r] = 0.f; }

  const int lk = tid & 63, cg = tid >> 6;   // loader coords: key, 32-dim group

  for (int kt = 0; kt < SEQ / 64; ++kt) {
    const __bf16* kp = base + (size_t)(kt * 64 + lk) * 3072 + 1024 + h * HD + cg * 32;
    const __bf16* vp = base + (size_t)(kt * 64 + lk) * 3072 + 2048 + h * HD + cg * 32;
    // K tile via async global->LDS
    cp16_to_lds(&kp[0],  &Kt[lk][cg * 32 + 0]);
    cp16_to_lds(&kp[8],  &Kt[lk][cg * 32 + 8]);
    cp16_to_lds(&kp[16], &Kt[lk][cg * 32 + 16]);
    cp16_to_lds(&kp[24], &Kt[lk][cg * 32 + 24]);
    // V tile transposed via VALU (element transpose can't use async DMA)
    for (int i = 0; i < 32; ++i) Vt[cg * 32 + i][lk] = vp[i];
    async_wait_all();
    __syncthreads();

    // S = Q K^T : 4 key-tiles x 2 k-chunks = 8 WMMAs
    v8f sa[4];
    for (int j = 0; j < 4; ++j) {
      sa[j] = vz;
      for (int kc = 0; kc < 2; ++kc) {
        v16bf kf = *(const v16bf*)&Kt[j * 16 + ln][kc * 32 + 16 * hf];
        sa[j] = __builtin_amdgcn_wmma_f32_16x16x32_bf16(
            false, qf[kc], false, kf, (short)0, sa[j], false, false);
      }
    }

    // online softmax: row M = r + 8*hf lives across one 16-lane half
    for (int r = 0; r < 8; ++r) {
      float s0 = sa[0][r] * 0.125f, s1 = sa[1][r] * 0.125f;
      float s2 = sa[2][r] * 0.125f, s3 = sa[3][r] * 0.125f;
      float mx = fmaxf(fmaxf(s0, s1), fmaxf(s2, s3));
      for (int off = 1; off < 16; off <<= 1) mx = fmaxf(mx, __shfl_xor(mx, off, 32));
      float mnew  = fmaxf(mrow[r], mx);
      float alpha = __expf(mrow[r] - mnew);
      mrow[r] = mnew;
      float p0 = __expf(s0 - mnew), p1 = __expf(s1 - mnew);
      float p2 = __expf(s2 - mnew), p3 = __expf(s3 - mnew);
      int rr = 8 * hf + r;
      Pt[w][rr][ 0 + ln] = f2bf(p0);
      Pt[w][rr][16 + ln] = f2bf(p1);
      Pt[w][rr][32 + ln] = f2bf(p2);
      Pt[w][rr][48 + ln] = f2bf(p3);
      float ps = p0 + p1 + p2 + p3;
      for (int off = 1; off < 16; off <<= 1) ps += __shfl_xor(ps, off, 32);
      lrow[r] = lrow[r] * alpha + ps;
      for (int j = 0; j < 4; ++j) o[j][r] *= alpha;
    }
    // same-wave LDS write->read of Pt: make the dependency explicit
    asm volatile("s_wait_dscnt 0" ::: "memory");

    // O += P V : P back through LDS in A-layout, V transposed for B-layout
    for (int kc = 0; kc < 2; ++kc) {
      U16 pa;
      pa.h[0] = *(const v8bf*)&Pt[w][ln][kc * 32 + 16 * hf];
      pa.h[1] = *(const v8bf*)&Pt[w][ln][kc * 32 + 16 * hf + 8];
      for (int j = 0; j < 4; ++j) {
        v16bf vf = *(const v16bf*)&Vt[j * 16 + ln][kc * 32 + 16 * hf];
        o[j] = __builtin_amdgcn_wmma_f32_16x16x32_bf16(
            false, pa.v, false, vf, (short)0, o[j], false, false);
      }
    }
    __syncthreads();
  }

  for (int j = 0; j < 4; ++j)
    for (int r = 0; r < 8; ++r) {
      int s = q0 + 8 * hf + r;
      float val = o[j][r] / lrow[r];
      out[((size_t)(b * SEQ + s)) * D_MODEL + h * HD + j * 16 + ln] = f2bf(val);
    }
}

// ---------------------------------------------------------------------------
// Host-side orchestration.
// ---------------------------------------------------------------------------
extern "C" void kernel_launch(void* const* d_in, const int* in_sizes, int n_in,
                              void* d_out, int out_size, void* d_ws, size_t ws_size,
                              hipStream_t stream) {
  (void)in_sizes; (void)n_in; (void)out_size; (void)ws_size;
  const float* x       = (const float*)d_in[0];
  const float* w_in_l  = (const float*)d_in[1];
  const float* b_in_l  = (const float*)d_in[2];
  const float* w_out_l = (const float*)d_in[3];
  const float* b_out_l = (const float*)d_in[4];
  const float* w_in_g  = (const float*)d_in[5];
  const float* b_in_g  = (const float*)d_in[6];
  const float* w_out_g = (const float*)d_in[7];
  const float* b_out_g = (const float*)d_in[8];

  // workspace carve-out (~64 MiB total)
  char* ws = (char*)d_ws;
  size_t off = 0;
  auto carve = [&](size_t bytes) {
    char* p = ws + off;
    off += (bytes + 255) & ~(size_t)255;
    return p;
  };
  __bf16* x_perm = (__bf16*)carve((size_t)NTOK * D_MODEL * 2);
  __bf16* qkv    = (__bf16*)carve((size_t)NTOK * 3 * D_MODEL * 2);
  __bf16* attn   = (__bf16*)carve((size_t)NTOK * D_MODEL * 2);
  __bf16* loc    = (__bf16*)carve((size_t)NTOK * D_MODEL * 2);
  __bf16* wil    = (__bf16*)carve((size_t)3 * D_MODEL * D_MODEL * 2);
  __bf16* wol    = (__bf16*)carve((size_t)D_MODEL * D_MODEL * 2);
  __bf16* wig    = (__bf16*)carve((size_t)3 * D_MODEL * D_MODEL * 2);
  __bf16* wog    = (__bf16*)carve((size_t)D_MODEL * D_MODEL * 2);

  const int nW3 = 3 * D_MODEL * D_MODEL;  // 3,145,728
  const int nW1 = D_MODEL * D_MODEL;      // 1,048,576
  cast_f32_bf16<<<nW3 / 256, 256, 0, stream>>>(w_in_l,  wil, nW3);
  cast_f32_bf16<<<nW1 / 256, 256, 0, stream>>>(w_out_l, wol, nW1);
  cast_f32_bf16<<<nW3 / 256, 256, 0, stream>>>(w_in_g,  wig, nW3);
  cast_f32_bf16<<<nW1 / 256, 256, 0, stream>>>(w_out_g, wog, nW1);
  permute_cast_x<<<(NTOK * D_MODEL) / 256, 256, 0, stream>>>(x, x_perm);

  // local stage
  gemm_bf16<1, 0><<<dim3(3 * D_MODEL / 256, NTOK / 64), 256, 0, stream>>>(
      x_perm, wil, b_in_l, qkv, NTOK, 3 * D_MODEL, D_MODEL);
  local_attn<<<(BATCH * CS * NHEADS) / 256, 256, 0, stream>>>(qkv, attn);
  gemm_bf16<1, 1><<<dim3(D_MODEL / 256, NTOK / 64), 256, 0, stream>>>(
      attn, wol, b_out_l, loc, NTOK, D_MODEL, D_MODEL);   // row-permute epilogue

  // global stage
  gemm_bf16<1, 0><<<dim3(3 * D_MODEL / 256, NTOK / 64), 256, 0, stream>>>(
      loc, wig, b_in_g, qkv, NTOK, 3 * D_MODEL, D_MODEL);
  global_attn<<<dim3(SEQ / 64, BATCH * NHEADS), 128, 0, stream>>>(qkv, attn);
  gemm_bf16<0, 0><<<dim3(D_MODEL / 256, NTOK / 64), 256, 0, stream>>>(
      attn, wog, b_out_g, d_out, NTOK, D_MODEL, D_MODEL);
}
